// SessionGraphAttn_9113920602382
// MI455X (gfx1250) — matrix operand order repaired and checked by reading, not compiled, not run on hardware
//
#include <hip/hip_runtime.h>
#include <hip/hip_bf16.h>
#include <math.h>

#define BB 512
#define LL 70
#define DD 128
#define HH 8
#define LT 10          // L-tile for the q2/sigmoid stage (70 = 7*10)

#define NCOLS 99999    // emb[1:]
#define NROWPAD 100000 // padded to 6250 * 16
#define NCT 6250       // column tiles of 16
#define NRT 32         // row tiles of 16 (512/16)

typedef __attribute__((ext_vector_type(16))) _Float16 v16h;
typedef __attribute__((ext_vector_type(8)))  _Float16 v8h;
typedef __attribute__((ext_vector_type(8)))  float    v8f;

// ---------------------------------------------------------------------------
// Kernel 1: per-batch graph -> normalized, 12x-scaled query vector (f16)
// ---------------------------------------------------------------------------
__global__ __launch_bounds__(128) void prep_kernel(
    const float* __restrict__ hidden, const float* __restrict__ hidden0,
    const int*   __restrict__ mask,   const float* __restrict__ s,
    const float* __restrict__ attn_in_w, const float* __restrict__ attn_in_b,
    const float* __restrict__ attn_out_w, const float* __restrict__ attn_out_b,
    const float* __restrict__ w_zero, const float* __restrict__ b_zero,
    const float* __restrict__ w_one,  const float* __restrict__ b_one,
    const float* __restrict__ w_two,  const float* __restrict__ b_two,
    const float* __restrict__ w_three, const float* __restrict__ w_gate,
    const float* __restrict__ w_trans, const float* __restrict__ b_trans,
    _Float16* __restrict__ a16)
{
    __shared__ float sh_h[LL][DD];      // hidden, later gated hidden (35.8 KB)
    __shared__ float sh_hk[HH][DD];     // q^T Wk per head, later hbar per head
    __shared__ float sh_tile[LT][DD];   // sigmoid(q0+q1+q2) tile
    __shared__ float sh_q[DD];
    __shared__ float sh_vec[DD];        // s row -> o -> a
    __shared__ float sh_ht[DD];
    __shared__ float sh_ht0[DD];
    __shared__ float sh_sc[HH][LL];     // MHA scores -> softmax weights
    __shared__ float sh_alpha[LL][HH];
    __shared__ float sh_red[DD];
    __shared__ float sh_sig[LL];
    __shared__ float sh_qb[HH];
    __shared__ int   sh_last;

    const int b = blockIdx.x;
    const int t = threadIdx.x;
    const float* hb  = hidden  + (size_t)b * LL * DD;
    const float* h0b = hidden0 + (size_t)b * LL * DD;

    // load hidden tile, last valid index
    for (int i = t; i < LL * DD; i += 128) ((float*)sh_h)[i] = hb[i];
    if (t == 0) {
        int m = 0;
        for (int l = 0; l < LL; ++l) m += mask[b * LL + l];
        int idx = m - 1; if (idx < 0) idx = 0; if (idx > LL - 1) idx = LL - 1;
        sh_last = idx;
    }
    sh_vec[t] = s[(size_t)b * DD + t];
    __syncthreads();
    sh_ht0[t] = sh_h[sh_last][t];

    // qp = s @ Wq^T + bq
    {
        float acc = attn_in_b[t];
        const float* wr = attn_in_w + (size_t)t * DD;
        for (int k = 0; k < DD; ++k) acc += sh_vec[k] * wr[k];
        sh_q[t] = acc;
    }
    __syncthreads();

    // per-head folded key projection: qk_h[k] = sum_j qp[h*16+j] * Wk[h*16+j][k]
    for (int task = t; task < HH * DD; task += 128) {
        int h = task >> 7, k = task & (DD - 1);
        float acc = 0.f;
        const float* wk = attn_in_w + (size_t)(DD + h * 16) * DD + k;
        #pragma unroll
        for (int j = 0; j < 16; ++j) acc += sh_q[h * 16 + j] * wk[(size_t)j * DD];
        sh_hk[h][k] = acc;
    }
    if (t < HH) {
        float acc = 0.f;
        #pragma unroll
        for (int j = 0; j < 16; ++j) acc += sh_q[t * 16 + j] * attn_in_b[DD + t * 16 + j];
        sh_qb[t] = acc;
    }
    __syncthreads();

    // scores[h][l] = 0.25 * (qb_h + qk_h . hidden[l])
    for (int task = t; task < HH * LL; task += 128) {
        int h = task / LL, l = task % LL;
        float acc = sh_qb[h];
        for (int k = 0; k < DD; ++k) acc += sh_hk[h][k] * sh_h[l][k];
        sh_sc[h][l] = acc * 0.25f;
    }
    __syncthreads();

    // softmax over l per head (no mask in MHA, per reference)
    if (t < HH) {
        float m = -3.4e38f;
        for (int l = 0; l < LL; ++l) m = fmaxf(m, sh_sc[t][l]);
        float sum = 0.f;
        for (int l = 0; l < LL; ++l) { float e = __expf(sh_sc[t][l] - m); sh_sc[t][l] = e; sum += e; }
        float inv = 1.f / sum;
        for (int l = 0; l < LL; ++l) sh_sc[t][l] *= inv;
    }
    __syncthreads();

    // hbar_h[k] = sum_l w[h][l] * hidden[l][k]   (overwrite sh_hk)
    for (int task = t; task < HH * DD; task += 128) {
        int h = task >> 7, k = task & (DD - 1);
        float acc = 0.f;
        for (int l = 0; l < LL; ++l) acc += sh_sc[h][l] * sh_h[l][k];
        sh_hk[h][k] = acc;
    }
    __syncthreads();

    // o[d] = bv[d] + Wv[d] . hbar_{d/16}  (softmax weights sum to 1)
    {
        int h = t >> 4;
        float acc = attn_in_b[2 * DD + t];
        const float* wv = attn_in_w + (size_t)(2 * DD + t) * DD;
        for (int k = 0; k < DD; ++k) acc += sh_hk[h][k] * wv[k];
        sh_vec[t] = acc;
    }
    __syncthreads();

    // ht = o @ Wout^T + bout
    {
        float acc = attn_out_b[t];
        const float* wr = attn_out_w + (size_t)t * DD;
        for (int k = 0; k < DD; ++k) acc += sh_vec[k] * wr[k];
        sh_ht[t] = acc;
    }
    __syncthreads();

    // q0 + q1 (per output dim t)
    float q01;
    {
        float acc1 = b_one[t];
        const float* w1 = w_one + (size_t)t * DD;
        for (int k = 0; k < DD; ++k) acc1 += sh_ht[k] * w1[k];
        float acc0 = b_zero[t];
        const float* w0 = w_zero + (size_t)t * DD;
        for (int k = 0; k < DD; ++k) acc0 += sh_ht0[k] * w0[k];
        q01 = acc0 + acc1;
    }

    // gate: sigma[l] = sigmoid([h0,h] . w_gate), hidden <- h0*sig + h*(1-sig)
    if (t < LL) {
        float acc = 0.f;
        const float* h0r = h0b + (size_t)t * DD;
        for (int d = 0; d < DD; ++d)
            acc += h0r[d] * w_gate[d] + sh_h[t][d] * w_gate[DD + d];
        sh_sig[t] = 1.f / (1.f + __expf(-acc));
    }
    __syncthreads();
    for (int l = 0; l < LL; ++l) {
        float g = sh_sig[l];
        sh_h[l][t] = h0b[(size_t)l * DD + t] * g + sh_h[l][t] * (1.f - g);
    }
    __syncthreads();

    // alpha[l][h] = sigmoid(q0+q1+q2[l]) . w_three[h], tiled over l
    for (int l0 = 0; l0 < LL; l0 += LT) {
        float acc[LT];
        float base = b_two[t] + q01;
        #pragma unroll
        for (int i = 0; i < LT; ++i) acc[i] = base;
        const float* w2 = w_two + (size_t)t * DD;
        for (int kb = 0; kb < DD; kb += 32) {
            float w[32];
            #pragma unroll
            for (int j = 0; j < 32; ++j) w[j] = w2[kb + j];
            #pragma unroll
            for (int i = 0; i < LT; ++i) {
                float a = acc[i];
                #pragma unroll
                for (int j = 0; j < 32; ++j) a += sh_h[l0 + i][kb + j] * w[j];
                acc[i] = a;
            }
        }
        #pragma unroll
        for (int i = 0; i < LT; ++i)
            sh_tile[i][t] = 1.f / (1.f + __expf(-acc[i]));
        __syncthreads();
        for (int task = t; task < LT * HH; task += 128) {
            int i = task >> 3, h = task & 7;
            float a = 0.f;
            const float* w3 = w_three + (size_t)h * DD;
            for (int d = 0; d < DD; ++d) a += sh_tile[i][d] * w3[d];
            sh_alpha[l0 + i][h] = a;
        }
        __syncthreads();
    }

    // masked softmax over l per head
    if (t < HH) {
        float m = -3.4e38f;
        for (int l = 0; l < LL; ++l)
            if (mask[b * LL + l]) m = fmaxf(m, sh_alpha[l][t]);
        float sum = 0.f;
        for (int l = 0; l < LL; ++l) {
            float e = mask[b * LL + l] ? __expf(sh_alpha[l][t] - m) : 0.f;
            sh_alpha[l][t] = e; sum += e;
        }
        float inv = (sum > 0.f) ? 1.f / sum : 0.f;
        for (int l = 0; l < LL; ++l) sh_alpha[l][t] *= inv;
    }
    __syncthreads();

    // a[d] = sum_l alpha[l][d/16] * gated[l][d] * mask[l]
    {
        int h = t >> 4;
        float acc = 0.f;
        for (int l = 0; l < LL; ++l) {
            float mf = (float)mask[b * LL + l];
            acc += sh_alpha[l][h] * sh_h[l][t] * mf;
        }
        sh_vec[t] = acc;
    }
    __syncthreads();

    // res = [a, ht0] @ w_trans^T + b_trans
    float res;
    {
        float acc = b_trans[t];
        const float* wt = w_trans + (size_t)t * (2 * DD);
        for (int k = 0; k < DD; ++k) acc += sh_vec[k] * wt[k];
        for (int k = 0; k < DD; ++k) acc += sh_ht0[k] * wt[DD + k];
        res = acc;
    }

    // normalize, fold in 12.0 scale, emit f16
    sh_red[t] = res * res;
    __syncthreads();
    for (int sstep = 64; sstep > 0; sstep >>= 1) {
        if (t < sstep) sh_red[t] += sh_red[t + sstep];
        __syncthreads();
    }
    float r = sqrtf(sh_red[0]) + 1e-12f;
    a16[(size_t)b * DD + t] = (_Float16)(res / r * 12.0f);
}

// ---------------------------------------------------------------------------
// Kernel 2: normalize emb[1:] rows -> f16, zero-pad to 100000 rows
// ---------------------------------------------------------------------------
__global__ __launch_bounds__(128) void emb_norm_kernel(
    const float* __restrict__ emb, _Float16* __restrict__ b16)
{
    const int r = blockIdx.x;        // 0 .. NROWPAD-1
    const int t = threadIdx.x;
    __shared__ float red[128];
    float v = 0.f;
    if (r < NCOLS) v = emb[(size_t)(r + 1) * DD + t];
    red[t] = v * v;
    __syncthreads();
    for (int sstep = 64; sstep > 0; sstep >>= 1) {
        if (t < sstep) red[t] += red[t + sstep];
        __syncthreads();
    }
    float nrm = sqrtf(red[0]) + 1e-12f;
    b16[(size_t)r * DD + t] = (_Float16)(v / nrm);
}

// ---------------------------------------------------------------------------
// Kernel 3: scores[512 x 99999] = a16 @ b16^T via v_wmma_f32_16x16x32_f16.
// Hand-unrolled ping-pong double buffer (NO dynamic indexing -> all
// fragments stay in VGPRs): tile t+1's 8 global_load_b128 are in flight
// while tile t's 4 WMMAs execute on resident registers.
// ---------------------------------------------------------------------------
__global__ __launch_bounds__(256) void score_wmma_kernel(
    const _Float16* __restrict__ a16, const _Float16* __restrict__ b16,
    float* __restrict__ out)
{
    const int lane = threadIdx.x & 31;
    const int wave = threadIdx.x >> 5;
    const int wavesPerRow = gridDim.x * 8;
    const int m0 = blockIdx.y * 16;
    const int l15 = lane & 15;
    const int hi  = lane >> 4;           // 0: lanes 0-15, 1: lanes 16-31

    // --- A stripe (16 rows x 128 K), contiguous K order (same as B), not the ISA table order:
    // lanes 0-15: row=lane, elems 0..15 = K 0..15
    // lanes 16-31: row=lane-16, elems 0..15 = K 16..31
    const _Float16* arow = a16 + (size_t)(m0 + l15) * DD;
    const int akoff = hi * 16;
    v16h afrag[4];
    #pragma unroll
    for (int ks = 0; ks < 4; ++ks) {
        v8h lo  = *(const v8h*)(arow + ks * 32 + akoff);
        v8h hiq = *(const v8h*)(arow + ks * 32 + 8 + akoff);
        v16h f;
        #pragma unroll
        for (int j = 0; j < 8; ++j) { f[j] = lo[j]; f[8 + j] = hiq[j]; }
        afrag[ks] = f;
    }

    // --- B layout (32x16): lane n holds col n0+n, K 0..15 (lanes 0-15)
    // or K 16..31 (lanes 16-31) — contiguous 16 halves of the emb row.
    const int bkoff = hi * 16;

    int ct = blockIdx.x * 8 + wave;
    if (ct >= NCT) return;

    auto loadB = [&](int tile, v16h (&dst)[4]) {
        const _Float16* brow = b16 + (size_t)(tile * 16 + l15) * DD + bkoff;
        #pragma unroll
        for (int ks = 0; ks < 4; ++ks)
            dst[ks] = *(const v16h*)(brow + ks * 32);
    };
    auto tileOp = [&](int tile, const v16h (&bfr)[4]) {
        v8f c = {};
        #pragma unroll
        for (int ks = 0; ks < 4; ++ks)
            c = __builtin_amdgcn_wmma_f32_16x16x32_f16(
                    false, afrag[ks], false, bfr[ks],
                    (short)0, c, false, false);
        const int n = tile * 16 + l15;
        if (n < NCOLS) {
            float* orow = out + (size_t)(m0 + hi * 8) * NCOLS + n;
            #pragma unroll
            for (int r = 0; r < 8; ++r) orow[(size_t)r * NCOLS] = c[r];
        }
    };
    auto prefetch = [&](int tile) {
        if (tile < NCT)
            __builtin_prefetch(b16 + (size_t)(tile * 16 + l15) * DD + bkoff, 0, 3);
    };

    v16h bufA[4], bufB[4];
    loadB(ct, bufA);
    prefetch(ct + wavesPerRow);

    for (;;) {
        const int ct1 = ct + wavesPerRow;
        if (ct1 >= NCT) { tileOp(ct, bufA); break; }
        loadB(ct1, bufB);                 // in flight during bufA's WMMAs
        prefetch(ct1 + wavesPerRow);
        tileOp(ct, bufA);

        const int ct2 = ct1 + wavesPerRow;
        if (ct2 >= NCT) { tileOp(ct1, bufB); break; }
        loadB(ct2, bufA);                 // in flight during bufB's WMMAs
        prefetch(ct2 + wavesPerRow);
        tileOp(ct1, bufB);

        ct = ct2;
    }
}

// ---------------------------------------------------------------------------
extern "C" void kernel_launch(void* const* d_in, const int* in_sizes, int n_in,
                              void* d_out, int out_size, void* d_ws, size_t ws_size,
                              hipStream_t stream)
{
    (void)in_sizes; (void)n_in; (void)out_size; (void)ws_size;
    const float* hidden     = (const float*)d_in[0];
    const float* hidden0    = (const float*)d_in[1];
    const int*   mask       = (const int*)  d_in[2];
    const float* s          = (const float*)d_in[3];
    const float* emb        = (const float*)d_in[4];
    const float* attn_in_w  = (const float*)d_in[5];
    const float* attn_in_b  = (const float*)d_in[6];
    const float* attn_out_w = (const float*)d_in[7];
    const float* attn_out_b = (const float*)d_in[8];
    const float* w_zero     = (const float*)d_in[9];
    const float* b_zero     = (const float*)d_in[10];
    const float* w_one      = (const float*)d_in[11];
    const float* b_one      = (const float*)d_in[12];
    const float* w_two      = (const float*)d_in[13];
    const float* b_two      = (const float*)d_in[14];
    const float* w_three    = (const float*)d_in[15];
    const float* w_gate     = (const float*)d_in[16];
    const float* w_trans    = (const float*)d_in[17];
    const float* b_trans    = (const float*)d_in[18];

    _Float16* a16 = (_Float16*)d_ws;                                   // 512*128 f16
    _Float16* b16 = (_Float16*)((char*)d_ws + (size_t)BB * DD * 2);    // 100000*128 f16
    float*    out = (float*)d_out;

    prep_kernel<<<BB, 128, 0, stream>>>(hidden, hidden0, mask, s,
        attn_in_w, attn_in_b, attn_out_w, attn_out_b,
        w_zero, b_zero, w_one, b_one, w_two, b_two,
        w_three, w_gate, w_trans, b_trans, a16);

    emb_norm_kernel<<<NROWPAD, 128, 0, stream>>>(emb, b16);

    score_wmma_kernel<<<dim3(98, NRT), 256, 0, stream>>>(a16, b16, out);
}
